// RelativeAttention_27479200759950
// MI455X (gfx1250) — compile-verified, matrix operand order checked by reading, not run on hardware
//
#include <hip/hip_runtime.h>

// Problem constants (from the reference)
#define BB   8
#define QQ   512
#define KVK  512
#define SZ   768
#define HH   12
#define DD   64
#define ZZ   512
#define HDIM 768

typedef __attribute__((ext_vector_type(16))) __bf16 v16bf;
typedef __attribute__((ext_vector_type(8)))  float  v8f;
typedef __attribute__((ext_vector_type(4)))  float  f32x4;
typedef __attribute__((ext_vector_type(8)))  __bf16 bf16x8;
typedef __attribute__((ext_vector_type(4)))  __bf16 bf16x4;

// ---------------------------------------------------------------------------
// WMMA helpers (CDNA5 wave32, v_wmma_f32_16x16x32_bf16)
// A fragment (16x32, 16-bit): lane L -> row M = L&15; K slots:
//   lanes 0-15 : {0..7, 16..23},  lanes 16-31 : {8..15, 24..31}
// B fragment loaded with the identical pattern from a row-major-along-K
// matrix implements  D = A * W^T  (both operands K-contiguous in memory).
// C/D fragment: VGPR r, lane L -> (M = r + (L>=16 ? 8 : 0), N = L&15).
// Fragments are built purely with vector loads + convertvector/shufflevector
// (no per-element extracts) so the loads lower to global_load_b128.
// ---------------------------------------------------------------------------

__device__ __forceinline__ v8f wmma_bf16(v16bf a, v16bf b, v8f c) {
  return __builtin_amdgcn_wmma_f32_16x16x32_bf16(false, a, false, b,
                                                 (short)0, c, false, false);
}

__device__ __forceinline__ v16bf concat16(bf16x8 lo, bf16x8 hi) {
  return __builtin_shufflevector(lo, hi, 0, 1, 2, 3, 4, 5, 6, 7,
                                 8, 9, 10, 11, 12, 13, 14, 15);
}

// fragment from fp32 row-major [row][k]: 4x b128 loads, vector cvt to bf16
__device__ __forceinline__ v16bf frag_from_f32(const float* __restrict__ base,
                                               int ld, int k0, int lane) {
  const int half = (lane >> 4) * 8;
  const int r    = lane & 15;
  const float* p = base + (size_t)r * ld + k0 + half;
  const f32x4 a0 = ((const f32x4*)p)[0];
  const f32x4 a1 = ((const f32x4*)p)[1];
  const f32x4 b0 = ((const f32x4*)(p + 16))[0];
  const f32x4 b1 = ((const f32x4*)(p + 16))[1];
  const bf16x4 c0 = __builtin_convertvector(a0, bf16x4);
  const bf16x4 c1 = __builtin_convertvector(a1, bf16x4);
  const bf16x4 c2 = __builtin_convertvector(b0, bf16x4);
  const bf16x4 c3 = __builtin_convertvector(b1, bf16x4);
  const bf16x8 lo = __builtin_shufflevector(c0, c1, 0, 1, 2, 3, 4, 5, 6, 7);
  const bf16x8 hi = __builtin_shufflevector(c2, c3, 0, 1, 2, 3, 4, 5, 6, 7);
  return concat16(lo, hi);
}

// fragment from bf16 row-major [row][k]: 2x b128 loads
__device__ __forceinline__ v16bf frag_from_bf16(const __bf16* __restrict__ base,
                                                int ld, int k0, int lane) {
  const int half = (lane >> 4) * 8;
  const int r    = lane & 15;
  const __bf16* p = base + (size_t)r * ld + k0 + half;
  const bf16x8 lo = ((const bf16x8*)p)[0];
  const bf16x8 hi = ((const bf16x8*)(p + 16))[0];
  return concat16(lo, hi);
}

// ---------------------------------------------------------------------------
// Kernel 0: fp32 matrix transpose via LDS (32x32 tiles, padded stride).
// Used once per Wpk/Wpq so the position projections can use the fully
// vectorized row-major GEMM path.  In: rows x cols.  Out: cols x rows.
// ---------------------------------------------------------------------------
__global__ void __launch_bounds__(256)
transpose_kernel(const float* __restrict__ In, float* __restrict__ Out,
                 int rows, int cols) {
  __shared__ float tile[32][33];
  const int ctiles = cols >> 5;
  const int bx = blockIdx.x % ctiles;
  const int by = blockIdx.x / ctiles;
  const int tx = threadIdx.x & 31;
  const int ty = threadIdx.x >> 5;       // 0..7
  const int c0 = bx << 5, r0 = by << 5;
#pragma unroll
  for (int i = 0; i < 4; ++i)
    tile[ty + 8 * i][tx] = In[(size_t)(r0 + ty + 8 * i) * cols + c0 + tx];
  __syncthreads();
#pragma unroll
  for (int i = 0; i < 4; ++i)
    Out[(size_t)(c0 + ty + 8 * i) * rows + r0 + tx] = tile[tx][ty + 8 * i];
}

// ---------------------------------------------------------------------------
// Kernel 1: projections  Out = X(MxSZ) * W(NxSZ)^T + bias, output bf16.
// OMODE 0: row-major (M x N)   (used for cq, ck, pk, pq)
// OMODE 1: per-head transposed cvT[(b*H+h)*D + d][k]  (used for cv)
// One wave = 16x64 output tile; 4 waves / block.
// ---------------------------------------------------------------------------
template <int OMODE>
__global__ void __launch_bounds__(128)
proj_kernel(const float* __restrict__ X, const float* __restrict__ W,
            const float* __restrict__ bias, __bf16* __restrict__ Out,
            int M, int N) {
  const int lane   = threadIdx.x & 31;
  const int waveId = blockIdx.x * 4 + (threadIdx.x >> 5);
  const int mtiles = M >> 4;
  const int m0 = (waveId % mtiles) << 4;
  const int n0 = (waveId / mtiles) << 6;
  if (n0 >= N) return;

  v8f acc[4] = {};
#pragma unroll 2
  for (int k0 = 0; k0 < SZ; k0 += 32) {
    v16bf a = frag_from_f32(X + (size_t)m0 * SZ, SZ, k0, lane);
#pragma unroll
    for (int j = 0; j < 4; ++j) {
      v16bf b = frag_from_f32(W + (size_t)(n0 + 16 * j) * SZ, SZ, k0, lane);
      acc[j] = wmma_bf16(a, b, acc[j]);
    }
  }

  const int col   = lane & 15;
  const int rbase = (lane >> 4) * 8;
#pragma unroll
  for (int j = 0; j < 4; ++j) {
    const int   n  = n0 + 16 * j + col;
    const float bv = bias[n];
#pragma unroll
    for (int r = 0; r < 8; ++r) {
      const int   m = m0 + rbase + r;
      const float v = acc[j][r] + bv;
      if (OMODE == 0) {
        Out[(size_t)m * N + n] = (__bf16)v;
      } else {
        const int bidx = m >> 9;   // m / 512 (K)
        const int kk   = m & 511;
        const int h    = n >> 6;   // n / 64
        const int d    = n & 63;
        Out[(((size_t)(bidx * HH + h) * DD + d) << 9) + kk] = (__bf16)v;
      }
    }
  }
}

// ---------------------------------------------------------------------------
// Kernel 3: per-(b,h) score GEMMs over D=64:  Out[m][n] = dot(Arow_m, Brow_n)
// A rows / B rows are bf16 with row stride HDIM (head slice offset h*D).
// Covers  S1 = cq*ck^T,  S2 = cq*pk^T,  S3 = pq*ck^T  (batch stride 0 for pos)
// ---------------------------------------------------------------------------
__global__ void __launch_bounds__(128)
head_gemm_kernel(const __bf16* __restrict__ A, size_t aBatchStride,
                 const __bf16* __restrict__ Bm, size_t bBatchStride,
                 float* __restrict__ Out, int M, int N) {
  const int bh = blockIdx.y;
  const int b  = bh / HH;
  const int h  = bh - b * HH;
  const __bf16* As = A  + (size_t)b * aBatchStride + h * DD;
  const __bf16* Bs = Bm + (size_t)b * bBatchStride + h * DD;
  float* Os = Out + (size_t)bh * M * N;

  const int lane   = threadIdx.x & 31;
  const int waveId = blockIdx.x * 4 + (threadIdx.x >> 5);
  const int mtiles = M >> 4;
  const int m0 = (waveId % mtiles) << 4;
  const int n0 = (waveId / mtiles) << 6;

  v8f acc[4] = {};
#pragma unroll
  for (int k0 = 0; k0 < DD; k0 += 32) {
    v16bf a = frag_from_bf16(As + (size_t)m0 * HDIM, HDIM, k0, lane);
#pragma unroll
    for (int j = 0; j < 4; ++j) {
      v16bf bb = frag_from_bf16(Bs + (size_t)(n0 + 16 * j) * HDIM, HDIM, k0, lane);
      acc[j] = wmma_bf16(a, bb, acc[j]);
    }
  }

  const int col   = lane & 15;
  const int rbase = (lane >> 4) * 8;
#pragma unroll
  for (int j = 0; j < 4; ++j)
#pragma unroll
    for (int r = 0; r < 8; ++r)
      Os[(size_t)(m0 + rbase + r) * N + n0 + 16 * j + col] = acc[j][r];
}

// ---------------------------------------------------------------------------
// Kernel 4: combine + softmax.  One block per (b,q); loops over H heads.
// attn[b,q,k,h] = (S1 + S2[gather z] + S3[gather z])*scale/sqrt(3) + head_bias
// weights (softmax over k with mask) written bf16 in (b,h,q,k) layout.
// Per-(b,h) S2/S3 slabs are 1MB -> gathers stay in L2 (192MB).
// attn is a pure streaming output -> nontemporal stores.
// ---------------------------------------------------------------------------
#define CTHREADS 256
__global__ void __launch_bounds__(CTHREADS)
combine_softmax_kernel(const float* __restrict__ S1, const float* __restrict__ S2,
                       const float* __restrict__ S3, const int* __restrict__ relpos,
                       const unsigned char* __restrict__ mask,
                       const float* __restrict__ head_bias,
                       float* __restrict__ attn, __bf16* __restrict__ Wgt) {
  const int b   = blockIdx.x / QQ;
  const int q   = blockIdx.x - b * QQ;
  const int tid = threadIdx.x;
  __shared__ float vals[KVK];
  __shared__ float red[CTHREADS];

  const int* rp = relpos + (size_t)(b * QQ + q) * KVK;  // X == 1
  const float scl = 0.125f * 0.57735026919f;            // 1/sqrt(D) * 1/sqrt(3)

  for (int h = 0; h < HH; ++h) {
    const size_t s1base = ((size_t)(b * HH + h) * QQ + q) * (size_t)KVK;
    const size_t s2base = ((size_t)(b * HH + h) * QQ + q) * (size_t)ZZ;
    const size_t s3base = (size_t)(b * HH + h) * ZZ * (size_t)KVK;
    const float  hb     = head_bias[h];

    for (int k = tid; k < KVK; k += CTHREADS) {
      int z = rp[k] + ZZ / 2;
      z = z < 0 ? 0 : (z > ZZ - 1 ? ZZ - 1 : z);
      const float a = (S1[s1base + k] + S2[s2base + z] +
                       S3[s3base + (size_t)z * KVK + k]) * scl + hb;
      __builtin_nontemporal_store(a, &attn[((size_t)(b * QQ + q) * KVK + k) * HH + h]);
      vals[k] = mask[(size_t)b * KVK + k] ? a : -1.0e6f;
    }
    __syncthreads();

    float mx = -3.402823466e38f;
    for (int k = tid; k < KVK; k += CTHREADS) mx = fmaxf(mx, vals[k]);
    red[tid] = mx; __syncthreads();
    for (int s = CTHREADS / 2; s > 0; s >>= 1) {
      if (tid < s) red[tid] = fmaxf(red[tid], red[tid + s]);
      __syncthreads();
    }
    mx = red[0]; __syncthreads();

    float sm = 0.0f;
    for (int k = tid; k < KVK; k += CTHREADS) {
      const float e = __expf(vals[k] - mx);
      vals[k] = e;
      sm += e;
    }
    red[tid] = sm; __syncthreads();
    for (int s = CTHREADS / 2; s > 0; s >>= 1) {
      if (tid < s) red[tid] += red[tid + s];
      __syncthreads();
    }
    const float inv = 1.0f / red[0];

    const size_t wbase = ((size_t)(b * HH + h) * QQ + q) * (size_t)KVK;
    for (int k = tid; k < KVK; k += CTHREADS)
      Wgt[wbase + k] = (__bf16)(vals[k] * inv);
    __syncthreads();
  }
}

// ---------------------------------------------------------------------------
// Kernel 5: pooled = weights * cv.  Per (b,h): (512x512)*(512x64) with cvT
// stored d-major/k-contiguous so both operands are K-contiguous bf16.
// pooled is a pure streaming output -> nontemporal stores.
// ---------------------------------------------------------------------------
__global__ void __launch_bounds__(128)
pooled_kernel(const __bf16* __restrict__ Wgt, const __bf16* __restrict__ cvT,
              float* __restrict__ pooled) {
  const int bh = blockIdx.y;
  const int b  = bh / HH;
  const int h  = bh - b * HH;
  const __bf16* A  = Wgt + (size_t)bh * QQ * KVK;
  const __bf16* Bm = cvT + (size_t)bh * DD * KVK;

  const int lane   = threadIdx.x & 31;
  const int waveId = blockIdx.x * 4 + (threadIdx.x >> 5);
  const int m0 = waveId << 4;

  v8f acc[4] = {};
#pragma unroll 4
  for (int k0 = 0; k0 < KVK; k0 += 32) {
    v16bf a = frag_from_bf16(A + (size_t)m0 * KVK, KVK, k0, lane);
#pragma unroll
    for (int j = 0; j < 4; ++j) {
      v16bf bb = frag_from_bf16(Bm + (size_t)(16 * j) * KVK, KVK, k0, lane);
      acc[j] = wmma_bf16(a, bb, acc[j]);
    }
  }

  const int col   = lane & 15;
  const int rbase = (lane >> 4) * 8;
#pragma unroll
  for (int j = 0; j < 4; ++j)
#pragma unroll
    for (int r = 0; r < 8; ++r) {
      const int q = m0 + rbase + r;
      const int d = 16 * j + col;
      __builtin_nontemporal_store(acc[j][r],
          &pooled[(size_t)(b * QQ + q) * HDIM + h * DD + d]);
    }
}

// ---------------------------------------------------------------------------
extern "C" void kernel_launch(void* const* d_in, const int* in_sizes, int n_in,
                              void* d_out, int out_size, void* d_ws, size_t ws_size,
                              hipStream_t stream) {
  const float*         cqs       = (const float*)d_in[0];
  const float*         cks       = (const float*)d_in[1];
  const float*         cvs       = (const float*)d_in[2];
  const unsigned char* mask      = (const unsigned char*)d_in[3];
  const int*           relpos    = (const int*)d_in[4];
  const float*         Wq        = (const float*)d_in[5];
  const float*         bq        = (const float*)d_in[6];
  const float*         Wk        = (const float*)d_in[7];
  const float*         bk        = (const float*)d_in[8];
  const float*         Wv        = (const float*)d_in[9];
  const float*         bv        = (const float*)d_in[10];
  const float*         Wpk       = (const float*)d_in[11];
  const float*         bpk       = (const float*)d_in[12];
  const float*         Wpq       = (const float*)d_in[13];
  const float*         bpq       = (const float*)d_in[14];
  const float*         pos_emb   = (const float*)d_in[15];
  const float*         head_bias = (const float*)d_in[16];

  char* ws = (char*)d_ws;
  size_t off = 0;
  auto wsalloc = [&](size_t bytes) -> char* {
    char* p = ws + off;
    off += (bytes + 255) & ~(size_t)255;
    return p;
  };
  __bf16* cq_bf  = (__bf16*)wsalloc((size_t)BB * QQ * HDIM * 2);
  __bf16* ck_bf  = (__bf16*)wsalloc((size_t)BB * KVK * HDIM * 2);
  __bf16* cvT_bf = (__bf16*)wsalloc((size_t)BB * HH * DD * KVK * 2);
  __bf16* pk_bf  = (__bf16*)wsalloc((size_t)ZZ * HDIM * 2);
  __bf16* pq_bf  = (__bf16*)wsalloc((size_t)ZZ * HDIM * 2);
  float*  WpkT   = (float*)wsalloc((size_t)SZ * HDIM * 4);
  float*  WpqT   = (float*)wsalloc((size_t)SZ * HDIM * 4);
  float*  S1     = (float*)wsalloc((size_t)BB * HH * QQ * KVK * 4);
  float*  S2     = (float*)wsalloc((size_t)BB * HH * QQ * ZZ * 4);
  float*  S3     = (float*)wsalloc((size_t)BB * HH * ZZ * KVK * 4);
  __bf16* Wgt    = (__bf16*)wsalloc((size_t)BB * HH * QQ * KVK * 2);

  float* pooled = (float*)d_out;
  float* attn   = pooled + (size_t)BB * QQ * HDIM;

  const dim3 blk(128);

  // 0) transpose position-projection weights (SZ x HDIM -> HDIM x SZ)
  {
    const int blocks = (SZ / 32) * (HDIM / 32);
    transpose_kernel<<<dim3(blocks), dim3(256), 0, stream>>>(Wpk, WpkT, SZ, HDIM);
    transpose_kernel<<<dim3(blocks), dim3(256), 0, stream>>>(Wpq, WpqT, SZ, HDIM);
  }
  // 1) content projections (M=4096, N=768, K=768)
  {
    const int M = BB * QQ, N = HDIM;
    const int blocks = ((M / 16) * (N / 64)) / 4;
    proj_kernel<0><<<dim3(blocks), blk, 0, stream>>>(cqs, Wq, bq, cq_bf, M, N);
    proj_kernel<0><<<dim3(blocks), blk, 0, stream>>>(cks, Wk, bk, ck_bf, M, N);
    proj_kernel<1><<<dim3(blocks), blk, 0, stream>>>(cvs, Wv, bv, cvT_bf, M, N);
  }
  // 2) position projections (M=512, N=768, K=768) using transposed weights
  {
    const int M = ZZ, N = HDIM;
    const int blocks = ((M / 16) * (N / 64)) / 4;
    proj_kernel<0><<<dim3(blocks), blk, 0, stream>>>(pos_emb, WpkT, bpk, pk_bf, M, N);
    proj_kernel<0><<<dim3(blocks), blk, 0, stream>>>(pos_emb, WpqT, bpq, pq_bf, M, N);
  }
  // 3) per-(b,h) 512x512x64 score GEMMs
  {
    const int blocks = ((QQ / 16) * (KVK / 64)) / 4;
    const dim3 grid(blocks, BB * HH);
    head_gemm_kernel<<<grid, blk, 0, stream>>>(cq_bf, (size_t)QQ * HDIM,
                                               ck_bf, (size_t)KVK * HDIM, S1, QQ, KVK);
    head_gemm_kernel<<<grid, blk, 0, stream>>>(cq_bf, (size_t)QQ * HDIM,
                                               pk_bf, (size_t)0, S2, QQ, ZZ);
    head_gemm_kernel<<<grid, blk, 0, stream>>>(pq_bf, (size_t)0,
                                               ck_bf, (size_t)KVK * HDIM, S3, ZZ, KVK);
  }
  // 4) gather + combine + softmax
  combine_softmax_kernel<<<dim3(BB * QQ), dim3(CTHREADS), 0, stream>>>(
      S1, S2, S3, relpos, mask, head_bias, attn, Wgt);
  // 5) pooled = weights * cv
  pooled_kernel<<<dim3((QQ / 16) / 4, BB * HH), blk, 0, stream>>>(Wgt, cvT_bf, pooled);
}
